// ReactionGCN_88991722373623
// MI455X (gfx1250) — matrix operand order repaired and checked by reading, not compiled, not run on hardware
//
#include <hip/hip_runtime.h>
#include <hip/hip_bf16.h>

// ---------------------------------------------------------------------------
// ReactionGCN on MI455X (gfx1250, wave32, WMMA).
//
// Roofline: per layer, the edge scatter moves ~1.6 GB (gather h[src] rows +
// atomic adds into agg[dst]) vs 3.3 GFLOP of GEMM. At 23.3 TB/s HBM the whole
// pipeline is memory/atomic bound (~0.3 ms), so the GEMM just needs to stay
// off the critical path: one wave = one 16x16 f32 tile via
// v_wmma_f32_16x16x32_f16, K=128 in 4 WMMA issues, weights pre-packed into
// the WMMA B lane layout for contiguous 128-bit loads.
// ---------------------------------------------------------------------------

typedef __attribute__((ext_vector_type(16))) _Float16 v16h;
typedef __attribute__((ext_vector_type(8)))  float    v8f;

#define F 128
#define EPS 1e-5f

// ---------------- prep: pack W[l] (f32 row-major KxN) into WMMA-B f16 layout
// Packed layout: [layer][coltile 8][kstep 4][lane 32][16 f16]
// lane -> N = ct*16 + (lane&15); elem j -> K = ks*32 + (lane>>4)*8 + (j&7) + (j>>3)*16
__global__ __launch_bounds__(256) void pack_w_kernel(const float* __restrict__ W,
                                                     _Float16* __restrict__ Wp,
                                                     int total) {
    int idx = blockIdx.x * 256 + threadIdx.x;
    if (idx >= total) return;
    int j    = idx & 15;
    int lane = (idx >> 4) & 31;
    int ks   = (idx >> 9) & 3;
    int ct   = (idx >> 11) & 7;
    int l    = idx >> 14;
    int n = ct * 16 + (lane & 15);
    int k = ks * 32 + (lane >> 4) * 8 + (j & 7) + (j >> 3) * 16;
    Wp[idx] = (_Float16)W[((size_t)l * F + k) * F + n];
}

// ---------------- f32 -> f16 activation conversion
__global__ __launch_bounds__(256) void cvt_f16_kernel(const float* __restrict__ in,
                                                      _Float16* __restrict__ out,
                                                      size_t total) {
    size_t i = blockIdx.x * (size_t)256 + threadIdx.x;
    if (i < total) out[i] = (_Float16)in[i];
}

// ---------------- degree + normalization
__global__ __launch_bounds__(256) void deg_kernel(const int* __restrict__ dst,
                                                  float* __restrict__ deg, int E) {
    int e = blockIdx.x * 256 + threadIdx.x;
    if (e < E) atomicAdd(&deg[dst[e]], 1.0f);
}

__global__ __launch_bounds__(256) void dis_kernel(const float* __restrict__ deg,
                                                  float* __restrict__ dis, int N) {
    int n = blockIdx.x * 256 + threadIdx.x;
    if (n < N) dis[n] = rsqrtf(deg[n] + 1.0f);
}

// ---------------- GEMM: h[N,128] = xh[N,128] @ Wl (packed), f32 out
// Block = 256 threads = 8 waves; wave w computes cols [16w,16w+16) of 16 rows.
__global__ __launch_bounds__(256) void gemm_kernel(const _Float16* __restrict__ xh,
                                                   const _Float16* __restrict__ Wp,
                                                   float* __restrict__ h) {
    const int wave = threadIdx.x >> 5;   // col tile 0..7
    const int lane = threadIdx.x & 31;
    const int m    = lane & 15;
    const int half = lane >> 4;
    const int row0 = blockIdx.x * 16;

    const _Float16* xrow = xh + (size_t)(row0 + m) * F;
    const _Float16* wp   = Wp + (size_t)wave * (4 * 32 * 16);

    union V16 { v16h v; uint4 q[2]; };
    v8f acc = {};
#pragma unroll
    for (int ks = 0; ks < 4; ++ks) {
        V16 a, b;
        // A: row m, K = ks*32 + half*8 + {0..7}  and  +16
        a.q[0] = *(const uint4*)(xrow + ks * 32 + half * 8);
        a.q[1] = *(const uint4*)(xrow + ks * 32 + 16 + half * 8);
        // B: pre-packed, contiguous per lane
        const _Float16* bp = wp + ((size_t)ks * 32 + lane) * 16;
        b.q[0] = *(const uint4*)(bp);
        b.q[1] = *(const uint4*)(bp + 8);
        acc = __builtin_amdgcn_wmma_f32_16x16x32_f16(
            /*neg_a=*/false, a.v, /*neg_b=*/false, b.v,
            /*c_mod=*/(short)0, acc, /*reuse_a=*/false, /*reuse_b=*/false);
    }
    // D: lane holds col m; VGPR r -> row r + 8*half
    float* hp = h + (size_t)(row0 + 8 * half) * F + wave * 16 + m;
#pragma unroll
    for (int r = 0; r < 8; ++r) hp[(size_t)r * F] = acc[r];
}

// ---------------- edge scatter: agg[dst] += h[src] * dis[src]*dis[dst]
// 32 threads per edge, 4 floats each -> coalesced 512B row gather.
__global__ __launch_bounds__(256) void scatter_kernel(const float* __restrict__ h,
                                                      const int* __restrict__ src,
                                                      const int* __restrict__ dst,
                                                      const float* __restrict__ dis,
                                                      float* __restrict__ agg, int E) {
    size_t t = blockIdx.x * (size_t)256 + threadIdx.x;
    size_t e = t >> 5;
    if (e >= (size_t)E) return;
    int fg = (int)(t & 31) * 4;
    int s = src[e], d = dst[e];
    float w = dis[s] * dis[d];
    const float4 hv = *(const float4*)(h + (size_t)s * F + fg);
    float* ap = agg + (size_t)d * F + fg;
    atomicAdd(ap + 0, hv.x * w);
    atomicAdd(ap + 1, hv.y * w);
    atomicAdd(ap + 2, hv.z * w);
    atomicAdd(ap + 3, hv.w * w);
}

// ---------------- fused: act = relu(agg + h*dis^2 + b), accumulate BN stats.
// 128 threads/block, thread owns feature f, in-place act overwrite of agg.
__global__ __launch_bounds__(128) void post_kernel(float* __restrict__ agg_act,
                                                   const float* __restrict__ h,
                                                   const float* __restrict__ dis,
                                                   const float* __restrict__ bias,
                                                   float* __restrict__ stats,
                                                   int N, int rowsPerBlock) {
    int f = threadIdx.x;
    int r0 = blockIdx.x * rowsPerBlock;
    int r1 = min(r0 + rowsPerBlock, N);
    float bf = bias[f];
    float s = 0.0f, sq = 0.0f;
    for (int r = r0; r < r1; ++r) {
        float sn = dis[r]; sn *= sn;
        size_t idx = (size_t)r * F + f;
        float v = agg_act[idx] + h[idx] * sn + bf;
        v = fmaxf(v, 0.0f);
        agg_act[idx] = v;
        s += v; sq += v * v;
    }
    atomicAdd(&stats[f], s);
    atomicAdd(&stats[F + f], sq);
}

// ---------------- BN finalize: fold gamma/beta into scale/shift
__global__ __launch_bounds__(128) void bn_finalize_kernel(const float* __restrict__ stats,
                                                          const float* __restrict__ gamma,
                                                          const float* __restrict__ beta,
                                                          float* __restrict__ scale,
                                                          float* __restrict__ shift,
                                                          float invN) {
    int f = threadIdx.x;
    float mu  = stats[f] * invN;
    float var = stats[F + f] * invN - mu * mu;
    float inv = rsqrtf(var + EPS);
    float sc  = gamma[f] * inv;
    scale[f] = sc;
    shift[f] = beta[f] - sc * mu;
}

// ---------------- normalize: x = act*scale + shift -> f32 (pool) + f16 (gemm)
__global__ __launch_bounds__(256) void norm_kernel(const float* __restrict__ act,
                                                   const float* __restrict__ scale,
                                                   const float* __restrict__ shift,
                                                   float* __restrict__ xf,
                                                   _Float16* __restrict__ xh,
                                                   size_t total) {
    size_t i = blockIdx.x * (size_t)256 + threadIdx.x;
    if (i >= total) return;
    int f = (int)(i & (F - 1));
    float v = act[i] * scale[f] + shift[f];
    xf[i] = v;
    xh[i] = (_Float16)v;
}

// ---------------- mean pool: batch is sorted -> run-length accumulate
__global__ __launch_bounds__(128) void pool_kernel(const float* __restrict__ xf,
                                                   const int* __restrict__ batch,
                                                   float* __restrict__ pooled,
                                                   float* __restrict__ counts,
                                                   int N, int rowsPerBlock) {
    int f = threadIdx.x;
    int r0 = blockIdx.x * rowsPerBlock;
    int r1 = min(r0 + rowsPerBlock, N);
    if (r0 >= r1) return;
    int curg = batch[r0];
    float acc = 0.0f, cnt = 0.0f;
    for (int r = r0; r < r1; ++r) {
        int g = batch[r];
        if (g != curg) {
            atomicAdd(&pooled[(size_t)curg * F + f], acc);
            if (f == 0) atomicAdd(&counts[curg], cnt);
            acc = 0.0f; cnt = 0.0f; curg = g;
        }
        acc += xf[(size_t)r * F + f];
        cnt += 1.0f;
    }
    atomicAdd(&pooled[(size_t)curg * F + f], acc);
    if (f == 0) atomicAdd(&counts[curg], cnt);
}

// ---------------- MLP head: out[g] = relu(pooled/cnt @ hw1 + hb1) @ hw2 + hb2
__global__ __launch_bounds__(64) void head_kernel(const float* __restrict__ pooled,
                                                  const float* __restrict__ counts,
                                                  const float* __restrict__ hw1,
                                                  const float* __restrict__ hb1,
                                                  const float* __restrict__ hw2,
                                                  const float* __restrict__ hb2,
                                                  float* __restrict__ out) {
    __shared__ float p[F];
    __shared__ float red[64];
    int g = blockIdx.x, j = threadIdx.x;
    float c = fmaxf(counts[g], 1.0f);
    p[j]      = pooled[(size_t)g * F + j] / c;
    p[j + 64] = pooled[(size_t)g * F + j + 64] / c;
    __syncthreads();
    float s = hb1[j];
#pragma unroll 8
    for (int k = 0; k < F; ++k) s += p[k] * hw1[k * 64 + j];
    s = fmaxf(s, 0.0f);
    red[j] = s * hw2[j];
    __syncthreads();
    for (int off = 32; off > 0; off >>= 1) {
        if (j < off) red[j] += red[j + off];
        __syncthreads();
    }
    if (j == 0) out[g] = red[0] + hb2[0];
}

extern "C" void kernel_launch(void* const* d_in, const int* in_sizes, int n_in,
                              void* d_out, int out_size, void* d_ws, size_t ws_size,
                              hipStream_t stream) {
    const float* x     = (const float*)d_in[0];
    const int*   ei    = (const int*)d_in[1];
    const int*   batch = (const int*)d_in[2];
    const float* W     = (const float*)d_in[3];
    const float* b     = (const float*)d_in[4];
    const float* gamma = (const float*)d_in[5];
    const float* beta  = (const float*)d_in[6];
    const float* hw1   = (const float*)d_in[7];
    const float* hb1   = (const float*)d_in[8];
    const float* hw2   = (const float*)d_in[9];
    const float* hb2   = (const float*)d_in[10];
    float* out = (float*)d_out;

    const int N = in_sizes[0] / F;      // 100000
    const int E = in_sizes[1] / 2;      // 1600000
    const int G = out_size;             // 2048
    const int L = 3;
    const int* src = ei;
    const int* dst = ei + E;

    // workspace carve-up (256B aligned)
    char* p = (char*)d_ws;
    auto alloc = [&](size_t bytes) -> void* {
        void* r = (void*)p;
        p += (bytes + 255) & ~(size_t)255;
        return r;
    };
    float*     deg    = (float*)alloc((size_t)N * 4);
    float*     dis    = (float*)alloc((size_t)N * 4);
    _Float16*  Wp     = (_Float16*)alloc((size_t)L * F * F * 2);
    _Float16*  xh     = (_Float16*)alloc((size_t)N * F * 2);
    float*     hbuf   = (float*)alloc((size_t)N * F * 4);   // GEMM out; final xf
    float*     agg    = (float*)alloc((size_t)N * F * 4);   // scatter out; act in-place
    float*     stats  = (float*)alloc(2 * F * 4);
    float*     scale  = (float*)alloc(F * 4);
    float*     shift  = (float*)alloc(F * 4);
    float*     pooled = (float*)alloc((size_t)G * F * 4);
    float*     counts = (float*)alloc((size_t)G * 4);
    (void)ws_size; (void)n_in;

    const size_t totNF = (size_t)N * F;
    const int STAT_BLOCKS = 512;
    const int rowsPerBlock = (N + STAT_BLOCKS - 1) / STAT_BLOCKS;

    // ---- prep
    {
        int totW = L * F * F;
        pack_w_kernel<<<(totW + 255) / 256, 256, 0, stream>>>(W, Wp, totW);
        cvt_f16_kernel<<<(int)((totNF + 255) / 256), 256, 0, stream>>>(x, xh, totNF);
        hipMemsetAsync(deg, 0, (size_t)N * 4, stream);
        deg_kernel<<<(E + 255) / 256, 256, 0, stream>>>(dst, deg, E);
        dis_kernel<<<(N + 255) / 256, 256, 0, stream>>>(deg, dis, N);
    }

    // ---- GCN layers
    for (int l = 0; l < L; ++l) {
        gemm_kernel<<<N / 16, 256, 0, stream>>>(xh, Wp + (size_t)l * F * F, hbuf);
        hipMemsetAsync(agg, 0, totNF * 4, stream);
        hipMemsetAsync(stats, 0, 2 * F * 4, stream);
        {
            size_t threads = (size_t)E * 32;
            scatter_kernel<<<(int)((threads + 255) / 256), 256, 0, stream>>>(
                hbuf, src, dst, dis, agg, E);
        }
        post_kernel<<<STAT_BLOCKS, 128, 0, stream>>>(agg, hbuf, dis, b + (size_t)l * F,
                                                     stats, N, rowsPerBlock);
        bn_finalize_kernel<<<1, 128, 0, stream>>>(stats, gamma + (size_t)l * F,
                                                  beta + (size_t)l * F,
                                                  scale, shift, 1.0f / (float)N);
        // writes xf into hbuf (free after post) and xh for next layer's GEMM
        norm_kernel<<<(int)((totNF + 255) / 256), 256, 0, stream>>>(
            agg, scale, shift, hbuf, xh, totNF);
    }

    // ---- pool + head
    hipMemsetAsync(pooled, 0, (size_t)G * F * 4, stream);
    hipMemsetAsync(counts, 0, (size_t)G * 4, stream);
    pool_kernel<<<STAT_BLOCKS, 128, 0, stream>>>(hbuf, batch, pooled, counts, N, rowsPerBlock);
    head_kernel<<<G, 64, 0, stream>>>(pooled, counts, hw1, hb1, hw2, hb2, out);
}
